// RelativeMultiHeadSelfAttentionModule_49855980372561
// MI455X (gfx1250) — matrix operand order repaired and checked by reading, not compiled, not run on hardware
//
#include <hip/hip_runtime.h>
#include <math.h>

// ---------------------------------------------------------------------------
// Problem constants (match reference)
// ---------------------------------------------------------------------------
#define BB 8
#define TT 1024
#define DD 512
#define HH 8
#define DK 64
#define NEG_BIG (-3.0e38f)

typedef __attribute__((ext_vector_type(16))) __bf16 v16bf;
typedef __attribute__((ext_vector_type(8)))  float  v8f;

__device__ __forceinline__ v8f wmma_bf16(v16bf a, v16bf b, v8f c) {
  // (neg_a, A, neg_b, B, c_mod, C, reuse_a, reuse_b)
  return __builtin_amdgcn_wmma_f32_16x16x32_bf16(false, a, false, b, (short)0, c,
                                                 false, false);
}

// ---------------------------------------------------------------------------
// WMMA fragment loaders (contiguous k = 16*half + s order, wave32)
// ---------------------------------------------------------------------------
// A matrix 16x32 (M x K), source row-major with leading dim ld.
// lane&15 = M row; lane>>4 selects K half. VGPR i holds K = 2i + 16*half.
__device__ __forceinline__ v16bf load_a_rowmajor(const __bf16* base, int ld) {
  int lane = threadIdx.x & 31;
  int m = lane & 15;
  int kh = (lane >> 4) * 16;
  const __bf16* row = base + (size_t)m * ld;
  v16bf a;
#pragma unroll
  for (int i = 0; i < 8; ++i) {
    int kb = 2 * i + kh;
    a[2 * i]     = row[kb];
    a[2 * i + 1] = row[kb + 1];
  }
  return a;
}

// B matrix 32x16 (K x N) where B[k][n] = W[n*ldw + k]  (W row-major [N][K]).
// lane&15 = N; koff = 16*(lane>>4); VGPR i holds K = koff + 2i, 2i+1.
// Per-lane data is 32 contiguous bytes -> two b128 loads.
__device__ __forceinline__ v16bf load_b_nmajor(const __bf16* W, int ldw) {
  int lane = threadIdx.x & 31;
  int n = lane & 15;
  int koff = (lane >> 4) * 16;
  const __bf16* row = W + (size_t)n * ldw + koff;
  v16bf b;
#pragma unroll
  for (int i = 0; i < 8; ++i) {
    b[2 * i]     = row[2 * i];
    b[2 * i + 1] = row[2 * i + 1];
  }
  return b;
}

// ---------------------------------------------------------------------------
// Kernel 0: f32 -> bf16 conversion (weights)
// ---------------------------------------------------------------------------
__global__ void cvt_kernel(const float* __restrict__ src, __bf16* __restrict__ dst, int n) {
  int i = blockIdx.x * 256 + threadIdx.x;
  if (i < n) dst[i] = (__bf16)src[i];
}

// ---------------------------------------------------------------------------
// Kernel 1: LayerNorm (one row of 512 per block of 256 threads) -> bf16
// ---------------------------------------------------------------------------
__global__ void ln_kernel(const float* __restrict__ x, const float* __restrict__ g,
                          const float* __restrict__ be, __bf16* __restrict__ xh) {
  int row = blockIdx.x;  // b*T + t
  const float* xr = x + (size_t)row * DD;
  float vals[2];
  float s = 0.f, s2 = 0.f;
#pragma unroll
  for (int i = 0; i < 2; ++i) {
    float v = xr[threadIdx.x + 256 * i];
    vals[i] = v; s += v; s2 += v * v;
  }
#pragma unroll
  for (int off = 1; off < 32; off <<= 1) {
    s  += __shfl_xor(s, off, 32);
    s2 += __shfl_xor(s2, off, 32);
  }
  __shared__ float sm[8], sm2[8];
  int wave = threadIdx.x >> 5;
  if ((threadIdx.x & 31) == 0) { sm[wave] = s; sm2[wave] = s2; }
  __syncthreads();
  float ts = 0.f, ts2 = 0.f;
#pragma unroll
  for (int i = 0; i < 8; ++i) { ts += sm[i]; ts2 += sm2[i]; }
  float mean = ts * (1.0f / DD);
  float var  = ts2 * (1.0f / DD) - mean * mean;
  float rstd = rsqrtf(var + 1e-5f);
  __bf16* out = xh + (size_t)row * DD;
#pragma unroll
  for (int i = 0; i < 2; ++i) {
    int c = threadIdx.x + 256 * i;
    out[c] = (__bf16)((vals[i] - mean) * rstd * g[c] + be[c]);
  }
}

// ---------------------------------------------------------------------------
// Kernel 2: p = pos_embed @ wp.T + bp   (B x 512, tiny -> VALU)
// ---------------------------------------------------------------------------
__global__ void pproj_kernel(const float* __restrict__ pos, const float* __restrict__ wp,
                             const float* __restrict__ bp, float* __restrict__ pOut) {
  int i = blockIdx.x * 256 + threadIdx.x;  // 0 .. B*512-1
  int b = i >> 9, n = i & 511;
  const float* pe = pos + (size_t)b * DD;
  const float* w  = wp + (size_t)n * DD;
  float s = bp[n];
  for (int k = 0; k < DD; ++k) s += pe[k] * w[k];
  pOut[i] = s;
}

// ---------------------------------------------------------------------------
// Kernel 3: fused QKV projection.  Each wave: 16 rows x 32 cols for q, k, v
// (A fragment loaded once per K-chunk, 6 WMMAs against it; 48 acc VGPRs ->
// no spill).  Grid = 512 t-tiles * 2 column groups = 1024 blocks of 256.
// q/k written head-major [B,H,T,DK]; v written TRANSPOSED [B,H,DK,T] so the
// attention P·V B-fragments become contiguous n-major loads.
// ---------------------------------------------------------------------------
__global__ void qkv_kernel(const __bf16* __restrict__ xh,
                           const __bf16* __restrict__ wqh, const __bf16* __restrict__ wkh,
                           const __bf16* __restrict__ wvh,
                           const float* __restrict__ bq, const float* __restrict__ bk,
                           const float* __restrict__ bv, const float* __restrict__ pbu,
                           __bf16* __restrict__ qu, __bf16* __restrict__ kbuf,
                           __bf16* __restrict__ vtbuf) {
  int wave = threadIdx.x >> 5;
  int lane = threadIdx.x & 31;
  int tIdx = blockIdx.x & 511;           // 512 t-tiles over B*T
  int nGrp = blockIdx.x >> 9;            // 0..1
  int t0 = tIdx * 16;                    // global row in [0, B*T)
  int n0 = (nGrp * 8 + wave) * 32;       // 32-wide column strip per wave

  v8f cq[2] = {v8f{}, v8f{}};
  v8f ck[2] = {v8f{}, v8f{}};
  v8f cv[2] = {v8f{}, v8f{}};
  const __bf16* abase = xh + (size_t)t0 * DD;
  const __bf16* arow  = abase + (size_t)(lane & 15) * DD;
  for (int kc = 0; kc < DD; kc += 32) {
    if (kc + 32 < DD) __builtin_prefetch(arow + kc + 32, 0, 1);  // global_prefetch_b8
    v16bf a = load_a_rowmajor(abase + kc, DD);
#pragma unroll
    for (int j = 0; j < 2; ++j) {
      size_t wb = (size_t)(n0 + j * 16) * DD + kc;
      cq[j] = wmma_bf16(a, load_b_nmajor(wqh + wb, DD), cq[j]);
      ck[j] = wmma_bf16(a, load_b_nmajor(wkh + wb, DD), ck[j]);
      cv[j] = wmma_bf16(a, load_b_nmajor(wvh + wb, DD), cv[j]);
    }
  }
  int b  = t0 >> 10;
  int tl = t0 & 1023;
  int m0 = (lane >> 4) * 8;
#pragma unroll
  for (int j = 0; j < 2; ++j) {
    int n  = n0 + j * 16 + (lane & 15);
    int h  = n >> 6, dd = n & 63;
    float biasq = bq[n] + pbu[n];  // pos_bias_u[h][dd] == pbu[n]
    float biask = bk[n];
    float biasv = bv[n];
    size_t qk_base = ((size_t)(b * HH + h) * TT + tl) * DK + dd;
    size_t vt_base = ((size_t)(b * HH + h) * DK + dd) * TT + tl;
#pragma unroll
    for (int r = 0; r < 8; ++r) {
      size_t o = qk_base + (size_t)(m0 + r) * DK;
      qu[o]   = (__bf16)(cq[j][r] + biasq);
      kbuf[o] = (__bf16)(ck[j][r] + biask);
      vtbuf[vt_base + m0 + r] = (__bf16)(cv[j][r] + biasv);  // contiguous in r
    }
  }
}

// ---------------------------------------------------------------------------
// Kernel 4: bd[b,h,t] = sum_d (qu - pbu + pbv) * p    (rank-1 positional term)
// ---------------------------------------------------------------------------
__global__ void bd_kernel(const __bf16* __restrict__ qu, const float* __restrict__ pOut,
                          const float* __restrict__ pbu, const float* __restrict__ pbv,
                          float* __restrict__ bdOut) {
  int i = blockIdx.x * 256 + threadIdx.x;  // (b*H+h)*T + t
  int bh = i >> 10;
  int b = bh >> 3, h = bh & 7;
  const __bf16* q = qu + (size_t)i * DK;
  const float* p = pOut + (size_t)b * DD + h * DK;
  const float* u = pbu + h * DK;
  const float* v = pbv + h * DK;
  float s = 0.f;
#pragma unroll 8
  for (int d = 0; d < DK; ++d) s += ((float)q[d] - u[d] + v[d]) * p[d];
  bdOut[i] = s;
}

// ---------------------------------------------------------------------------
// Kernel 5: flash attention per (b,h).  Block = 256 thr (8 waves), each wave
// owns 16 t-rows; grid = B*H*(T/128) = 512.  s streamed in steps of 32.
// V is consumed from the transposed [B,H,DK,T] layout via n-major loads.
// ---------------------------------------------------------------------------
__global__ void attn_kernel(const __bf16* __restrict__ qu, const __bf16* __restrict__ kbuf,
                            const __bf16* __restrict__ vtbuf, const float* __restrict__ bd,
                            const int* __restrict__ lens, __bf16* __restrict__ ctx) {
  __shared__ __bf16 pbuf[8][16 * 32];  // per-wave P tile (16 x 32)

  int wave = threadIdx.x >> 5;
  int lane = threadIdx.x & 31;
  int ncol = lane & 15;
  int m0   = (lane >> 4) * 8;

  int idx  = blockIdx.x;
  int tblk = idx & 7;
  int h    = (idx >> 3) & 7;
  int b    = idx >> 6;
  int t0   = tblk * 128 + wave * 16;

  size_t hbase = (size_t)(b * HH + h) * TT * DK;  // also == head base of vtbuf
  const __bf16* qbase = qu + hbase + (size_t)t0 * DK;
  v16bf aq0 = load_a_rowmajor(qbase, DK);
  v16bf aq1 = load_a_rowmajor(qbase + 32, DK);

  float bdrow[8];
#pragma unroll
  for (int r = 0; r < 8; ++r)
    bdrow[r] = bd[(size_t)(b * HH + h) * TT + t0 + m0 + r];

  int len = lens[b];

  float mi[8], li[8];
#pragma unroll
  for (int r = 0; r < 8; ++r) { mi[r] = NEG_BIG; li[r] = 0.f; }
  v8f acc[4] = {v8f{}, v8f{}, v8f{}, v8f{}};

  for (int s0 = 0; s0 < TT; s0 += 32) {
    if (s0 >= len) break;  // uniform across block (len depends only on b)

    const __bf16* kb = kbuf + hbase + (size_t)s0 * DK;
    v8f c0 = {};
    c0 = wmma_bf16(aq0, load_b_nmajor(kb, DK), c0);
    c0 = wmma_bf16(aq1, load_b_nmajor(kb + 32, DK), c0);
    v8f c1 = {};
    c1 = wmma_bf16(aq0, load_b_nmajor(kb + 16 * DK, DK), c1);
    c1 = wmma_bf16(aq1, load_b_nmajor(kb + 16 * DK + 32, DK), c1);

    bool v0 = (s0 + ncol) < len;
    bool v1 = (s0 + 16 + ncol) < len;
    float p0[8], p1[8], tmax[8];
#pragma unroll
    for (int r = 0; r < 8; ++r) {
      float x0 = v0 ? (c0[r] + bdrow[r]) * 0.125f : NEG_BIG;  // 1/sqrt(64)
      float x1 = v1 ? (c1[r] + bdrow[r]) * 0.125f : NEG_BIG;
      p0[r] = x0; p1[r] = x1;
      tmax[r] = fmaxf(x0, x1);
    }
#pragma unroll
    for (int off = 1; off < 16; off <<= 1)
#pragma unroll
      for (int r = 0; r < 8; ++r)
        tmax[r] = fmaxf(tmax[r], __shfl_xor(tmax[r], off, 32));

    float alpha[8], rs[8];
#pragma unroll
    for (int r = 0; r < 8; ++r) {
      float mn = fmaxf(mi[r], tmax[r]);
      alpha[r] = __expf(mi[r] - mn);
      mi[r] = mn;
      float e0 = __expf(p0[r] - mn);
      float e1 = __expf(p1[r] - mn);
      p0[r] = e0; p1[r] = e1;
      rs[r] = e0 + e1;
    }
#pragma unroll
    for (int off = 1; off < 16; off <<= 1)
#pragma unroll
      for (int r = 0; r < 8; ++r)
        rs[r] += __shfl_xor(rs[r], off, 32);
#pragma unroll
    for (int r = 0; r < 8; ++r) li[r] = li[r] * alpha[r] + rs[r];
#pragma unroll
    for (int j = 0; j < 4; ++j)
#pragma unroll
      for (int r = 0; r < 8; ++r) acc[j][r] *= alpha[r];

    // repack P (C layout f32) -> LDS row-major 16x32 bf16 -> A fragment
    __bf16* pb = &pbuf[wave][0];
#pragma unroll
    for (int r = 0; r < 8; ++r) {
      int m = m0 + r;
      pb[m * 32 + ncol]      = (__bf16)p0[r];
      pb[m * 32 + 16 + ncol] = (__bf16)p1[r];
    }
    __syncthreads();  // cross-lane exchange: fence compiler + waves (uniform trip count)
    v16bf ap = load_a_rowmajor(pb, 32);

    // P (16 x 32, K=s) x V (32 x 64) with V transposed: B[k][n] = vt[n*T + s0+k]
    const __bf16* vt = vtbuf + hbase;
#pragma unroll
    for (int j = 0; j < 4; ++j)
      acc[j] = wmma_bf16(ap, load_b_nmajor(vt + (size_t)(j * 16) * TT + s0, TT), acc[j]);
    __syncthreads();  // protect pbuf reads from next iteration's stores
  }

  // epilogue: out = acc / l, write ctx [B,T,D] bf16
#pragma unroll
  for (int j = 0; j < 4; ++j) {
#pragma unroll
    for (int r = 0; r < 8; ++r) {
      int t = t0 + m0 + r;
      float o = acc[j][r] / li[r];
      ctx[((size_t)(b * TT) + t) * DD + h * DK + j * 16 + ncol] = (__bf16)o;
    }
  }
}

// ---------------------------------------------------------------------------
// Kernel 6: output projection  out = ctx @ wo.T + bo   (f32 out)
// Each wave: 16 rows x 64 cols; block covers one t-tile; grid = 512.
// ---------------------------------------------------------------------------
__global__ void proj_kernel(const __bf16* __restrict__ ctx, const __bf16* __restrict__ woh,
                            const float* __restrict__ bo, float* __restrict__ out) {
  int wave = threadIdx.x >> 5;
  int lane = threadIdx.x & 31;
  int t0 = blockIdx.x * 16;
  int n0 = wave * 64;

  v8f c[4] = {v8f{}, v8f{}, v8f{}, v8f{}};
  const __bf16* abase = ctx + (size_t)t0 * DD;
  const __bf16* arow  = abase + (size_t)(lane & 15) * DD;
  for (int kc = 0; kc < DD; kc += 32) {
    if (kc + 32 < DD) __builtin_prefetch(arow + kc + 32, 0, 1);
    v16bf a = load_a_rowmajor(abase + kc, DD);
#pragma unroll
    for (int j = 0; j < 4; ++j)
      c[j] = wmma_bf16(a, load_b_nmajor(woh + (size_t)(n0 + j * 16) * DD + kc, DD), c[j]);
  }
  int m0 = (lane >> 4) * 8;
#pragma unroll
  for (int j = 0; j < 4; ++j) {
    int n = n0 + j * 16 + (lane & 15);
    float bias = bo[n];
#pragma unroll
    for (int r = 0; r < 8; ++r)
      out[(size_t)(t0 + m0 + r) * DD + n] = c[j][r] + bias;
  }
}

// ---------------------------------------------------------------------------
// Host launcher
// ---------------------------------------------------------------------------
extern "C" void kernel_launch(void* const* d_in, const int* in_sizes, int n_in,
                              void* d_out, int out_size, void* d_ws, size_t ws_size,
                              hipStream_t stream) {
  const float* inputs    = (const float*)d_in[0];
  const float* pos_embed = (const float*)d_in[1];
  const float* ln_gamma  = (const float*)d_in[2];
  const float* ln_beta   = (const float*)d_in[3];
  const float* wq        = (const float*)d_in[4];
  const float* bq        = (const float*)d_in[5];
  const float* wk        = (const float*)d_in[6];
  const float* bk        = (const float*)d_in[7];
  const float* wv        = (const float*)d_in[8];
  const float* bv        = (const float*)d_in[9];
  const float* wp        = (const float*)d_in[10];
  const float* bp        = (const float*)d_in[11];
  const float* wo        = (const float*)d_in[12];
  const float* bo        = (const float*)d_in[13];
  const float* pbu       = (const float*)d_in[14];
  const float* pbv       = (const float*)d_in[15];
  const int*   lens      = (const int*)d_in[16];
  float* out = (float*)d_out;

  char* ws = (char*)d_ws;
  const size_t SZ_ACT = (size_t)BB * TT * DD * sizeof(__bf16);  // 8 MB
  const size_t SZ_W   = (size_t)DD * DD * sizeof(__bf16);       // 512 KB
  __bf16* xh    = (__bf16*)(ws);
  __bf16* qu    = (__bf16*)(ws + SZ_ACT);
  __bf16* kbuf  = (__bf16*)(ws + 2 * SZ_ACT);
  __bf16* vtbuf = (__bf16*)(ws + 3 * SZ_ACT);
  __bf16* ctx   = (__bf16*)(ws + 4 * SZ_ACT);
  __bf16* wqh   = (__bf16*)(ws + 5 * SZ_ACT);
  __bf16* wkh   = (__bf16*)(ws + 5 * SZ_ACT + SZ_W);
  __bf16* wvh   = (__bf16*)(ws + 5 * SZ_ACT + 2 * SZ_W);
  __bf16* woh   = (__bf16*)(ws + 5 * SZ_ACT + 3 * SZ_W);
  float*  pOut  = (float*)(ws + 5 * SZ_ACT + 4 * SZ_W);
  float*  bdW   = (float*)(ws + 5 * SZ_ACT + 4 * SZ_W + (size_t)BB * DD * sizeof(float));

  const int NW = DD * DD;  // 262144 weight elements
  cvt_kernel<<<(NW + 255) / 256, 256, 0, stream>>>(wq, wqh, NW);
  cvt_kernel<<<(NW + 255) / 256, 256, 0, stream>>>(wk, wkh, NW);
  cvt_kernel<<<(NW + 255) / 256, 256, 0, stream>>>(wv, wvh, NW);
  cvt_kernel<<<(NW + 255) / 256, 256, 0, stream>>>(wo, woh, NW);

  ln_kernel<<<BB * TT, 256, 0, stream>>>(inputs, ln_gamma, ln_beta, xh);

  pproj_kernel<<<(BB * DD) / 256, 256, 0, stream>>>(pos_embed, wp, bp, pOut);

  qkv_kernel<<<1024, 256, 0, stream>>>(xh, wqh, wkh, wvh, bq, bk, bv, pbu,
                                       qu, kbuf, vtbuf);

  bd_kernel<<<(BB * HH * TT) / 256, 256, 0, stream>>>(qu, pOut, pbu, pbv, bdW);

  attn_kernel<<<BB * HH * (TT / 128), 256, 0, stream>>>(qu, kbuf, vtbuf, bdW, lens, ctx);

  proj_kernel<<<512, 256, 0, stream>>>(ctx, woh, bo, out);
}
